// _SelfAttention_78091095376074
// MI455X (gfx1250) — matrix operand order repaired and checked by reading, not compiled, not run on hardware
//
#include <hip/hip_runtime.h>
#include <hip/hip_bf16.h>

typedef __bf16 bf16x8  __attribute__((ext_vector_type(8)));
typedef __bf16 bf16x16 __attribute__((ext_vector_type(16)));
typedef float  f32x8   __attribute__((ext_vector_type(8)));

constexpr int Bn = 4, Sn = 2048, Dn = 1024, Hn = 16, HDn = 64;

__device__ __forceinline__ unsigned short f2bf_rne(float f) {
  union { float f; unsigned u; } v; v.f = f;
  unsigned u = v.u;
  u += 0x7FFFu + ((u >> 16) & 1u);           // round-to-nearest-even
  return (unsigned short)(u >> 16);
}

__device__ __forceinline__ bf16x16 cat8(bf16x8 lo, bf16x8 hi) {
  return __builtin_shufflevector(lo, hi, 0,1,2,3,4,5,6,7,8,9,10,11,12,13,14,15);
}

__device__ __forceinline__ f32x8 zero8() {
  f32x8 v;
#pragma unroll
  for (int i = 0; i < 8; i++) v[i] = 0.0f;
  return v;
}

// ---------------------------------------------------------------------------
// Kernel 1: QKV GEMM.  x[8192x1024] (f32) @ w_qkv[1024x3072] (f32) + bias.
// Output scattered as bf16:  Q,K -> [B,H,S,HD],  V -> transposed [B,H,HD,S].
// WG = 256 thr (8 waves), tile 128x128, K-step 32, wave tile 64x32 (8 wmma).
// ---------------------------------------------------------------------------
__global__ __launch_bounds__(256)
void qkv_gemm_kernel(const float* __restrict__ x, const float* __restrict__ wqkv,
                     const float* __restrict__ bqkv,
                     unsigned short* __restrict__ qws, unsigned short* __restrict__ kws,
                     unsigned short* __restrict__ vtws) {
  __shared__ unsigned short As[128 * 40];   // [row][k], stride 40 bf16 (80B)
  __shared__ unsigned short Bt[128 * 40];   // transposed: [n][k]
  const int tid  = threadIdx.x;
  const int lane = tid & 31;
  const int wave = tid >> 5;
  const int wm   = wave & 1;        // 0..1 -> 64-row half
  const int wn   = wave >> 1;       // 0..3 -> 32-col quarter
  const int m0   = blockIdx.x * 128;
  const int n0   = blockIdx.y * 128;

  f32x8 acc[4][2];
#pragma unroll
  for (int i = 0; i < 4; i++)
#pragma unroll
    for (int j = 0; j < 2; j++) acc[i][j] = zero8();

  const int arow = tid >> 3;            // 0..31
  const int acol = (tid & 7) << 2;      // 0..28 step 4
  const int bcol = tid & 127;
  const int bkh  = (tid >> 7) << 4;     // 0 or 16

  for (int k0 = 0; k0 < Dn; k0 += 32) {
    __syncthreads();
    // stage A (convert f32 -> bf16)
#pragma unroll
    for (int p = 0; p < 4; p++) {
      const int row = arow + p * 32;
      const float4 f = *reinterpret_cast<const float4*>(&x[(size_t)(m0 + row) * Dn + k0 + acol]);
      uint2 pk;
      pk.x = (unsigned)f2bf_rne(f.x) | ((unsigned)f2bf_rne(f.y) << 16);
      pk.y = (unsigned)f2bf_rne(f.z) | ((unsigned)f2bf_rne(f.w) << 16);
      *reinterpret_cast<uint2*>(&As[row * 40 + acol]) = pk;
    }
    // stage B transposed (convert f32 -> bf16)
    unsigned short tmp[16];
#pragma unroll
    for (int kr = 0; kr < 16; kr++)
      tmp[kr] = f2bf_rne(wqkv[(size_t)(k0 + bkh + kr) * (3 * Dn) + n0 + bcol]);
#pragma unroll
    for (int q = 0; q < 4; q++) {
      uint2 pk;
      pk.x = (unsigned)tmp[q * 4 + 0] | ((unsigned)tmp[q * 4 + 1] << 16);
      pk.y = (unsigned)tmp[q * 4 + 2] | ((unsigned)tmp[q * 4 + 3] << 16);
      *reinterpret_cast<uint2*>(&Bt[bcol * 40 + bkh + q * 4]) = pk;
    }
    __syncthreads();

    const int kb = (lane >> 4) << 4;   // A frag: 0 / 16
    const int ks = (lane >> 4) << 4;   // B frag: 0 / 16
    bf16x16 afr[4];
#pragma unroll
    for (int mt = 0; mt < 4; mt++) {
      const int row = wm * 64 + mt * 16 + (lane & 15);
      bf16x8 lo = *reinterpret_cast<const bf16x8*>(&As[row * 40 + kb]);
      bf16x8 hi = *reinterpret_cast<const bf16x8*>(&As[row * 40 + kb + 8]);
      afr[mt] = cat8(lo, hi);
    }
    bf16x16 bfr[2];
#pragma unroll
    for (int nt = 0; nt < 2; nt++) {
      const int col = wn * 32 + nt * 16 + (lane & 15);
      bf16x8 lo = *reinterpret_cast<const bf16x8*>(&Bt[col * 40 + ks]);
      bf16x8 hi = *reinterpret_cast<const bf16x8*>(&Bt[col * 40 + ks + 8]);
      bfr[nt] = cat8(lo, hi);
    }
#pragma unroll
    for (int mt = 0; mt < 4; mt++)
#pragma unroll
      for (int nt = 0; nt < 2; nt++)
        acc[mt][nt] = __builtin_amdgcn_wmma_f32_16x16x32_bf16(
            false, afr[mt], false, bfr[nt], (short)0, acc[mt][nt], false, false);
  }

  // epilogue: bias add, scatter into Q / K / Vt as bf16
  const int halfm = (lane >> 4) << 3;
#pragma unroll
  for (int nt = 0; nt < 2; nt++) {
    const int Ncol = n0 + wn * 32 + nt * 16 + (lane & 15);   // [0,3072)
    const float bias = bqkv[Ncol];
    const int which = Ncol >> 10;
    const int d = Ncol & 1023;
    const int h = d >> 6, hd = d & 63;
#pragma unroll
    for (int mt = 0; mt < 4; mt++) {
#pragma unroll
      for (int r = 0; r < 8; r++) {
        const int m = m0 + wm * 64 + mt * 16 + halfm + r;
        const int b = m >> 11, s = m & 2047;
        const unsigned short bv = f2bf_rne(acc[mt][nt][r] + bias);
        if (which == 0)
          qws[(size_t)((b * Hn + h) * Sn + s) * HDn + hd] = bv;
        else if (which == 1)
          kws[(size_t)((b * Hn + h) * Sn + s) * HDn + hd] = bv;
        else
          vtws[(size_t)((b * Hn + h) * HDn + hd) * Sn + s] = bv;
      }
    }
  }
}

// ---------------------------------------------------------------------------
// Kernel 2: flash attention.  One WG per (b,h,128 query rows); each wave owns
// 16 query rows, streams keys in chunks of 32 with online softmax.
// 8 wmma per key-chunk (4 QK^T + 4 PV).
// ---------------------------------------------------------------------------
__global__ __launch_bounds__(256)
void attn_kernel(const unsigned short* __restrict__ qws,
                 const unsigned short* __restrict__ kws,
                 const unsigned short* __restrict__ vtws,
                 unsigned short* __restrict__ aout) {
  __shared__ unsigned short P[8 * 16 * 40];   // per-wave 16x32 P tile, stride 40
  const int tid = threadIdx.x, lane = tid & 31, wave = tid >> 5;
  const int idx = blockIdx.x;
  const int b = idx >> 8, h = (idx >> 4) & 15, qt = idx & 15;
  const int bh = b * Hn + h;
  const int q0 = qt * 128 + wave * 16;

  const size_t kvbase = (size_t)bh * Sn * HDn;   // Q/K [S,HD]
  const size_t vtbase = (size_t)bh * HDn * Sn;   // Vt [HD,S]

  const int kb = (lane >> 4) << 4;       // A frag K offset: 0/16
  const int kscol = (lane >> 4) << 4;    // B frag K offset: 0/16
  const int lc = lane & 15;

  // persistent Q fragments (16 rows x 64 d = 2 frags)
  bf16x16 aQ[2];
  {
    const int qrow = q0 + lc;
#pragma unroll
    for (int ksg = 0; ksg < 2; ksg++) {
      const unsigned short* p = &qws[kvbase + (size_t)qrow * HDn + ksg * 32 + kb];
      bf16x8 lo = *reinterpret_cast<const bf16x8*>(p);
      bf16x8 hi = *reinterpret_cast<const bf16x8*>(p + 8);
      aQ[ksg] = cat8(lo, hi);
    }
  }

  float mi[8], li[8];
#pragma unroll
  for (int r = 0; r < 8; r++) { mi[r] = -3.0e38f; li[r] = 0.0f; }
  f32x8 accO[4];
#pragma unroll
  for (int nt = 0; nt < 4; nt++) accO[nt] = zero8();

  unsigned short* myP = &P[wave * 16 * 40];

  for (int j0 = 0; j0 < Sn; j0 += 32) {
    // scores: 16 queries x 32 keys
    f32x8 sc[2];
#pragma unroll
    for (int jt = 0; jt < 2; jt++) {
      sc[jt] = zero8();
      const int key = j0 + jt * 16 + lc;
#pragma unroll
      for (int ksg = 0; ksg < 2; ksg++) {
        const unsigned short* p = &kws[kvbase + (size_t)key * HDn + ksg * 32 + kscol];
        bf16x8 lo = *reinterpret_cast<const bf16x8*>(p);
        bf16x8 hi = *reinterpret_cast<const bf16x8*>(p + 8);
        bf16x16 bK = cat8(lo, hi);
        sc[jt] = __builtin_amdgcn_wmma_f32_16x16x32_bf16(
            false, aQ[ksg], false, bK, (short)0, sc[jt], false, false);
      }
    }
    // online softmax (row = r + 8*(lane>=16); reduce across the 16-lane group)
#pragma unroll
    for (int r = 0; r < 8; r++) {
      float s0 = sc[0][r] * 0.125f, s1 = sc[1][r] * 0.125f;  // 1/sqrt(64)
      float pm = fmaxf(s0, s1);
#pragma unroll
      for (int msk = 1; msk < 16; msk <<= 1) pm = fmaxf(pm, __shfl_xor(pm, msk, 32));
      const float mnew = fmaxf(mi[r], pm);
      const float p0 = __expf(s0 - mnew), p1 = __expf(s1 - mnew);
      float rs = p0 + p1;
#pragma unroll
      for (int msk = 1; msk < 16; msk <<= 1) rs += __shfl_xor(rs, msk, 32);
      const float alpha = __expf(mi[r] - mnew);
      li[r] = li[r] * alpha + rs;
      mi[r] = mnew;
#pragma unroll
      for (int nt = 0; nt < 4; nt++) accO[nt][r] *= alpha;
      const int M = r + ((lane >> 4) << 3);
      myP[M * 40 + lc]      = f2bf_rne(p0);
      myP[M * 40 + 16 + lc] = f2bf_rne(p1);
    }
    // wave-level LDS sync: all 32 lanes' ds_stores complete before re-reading
    asm volatile("s_wait_dscnt 0" ::: "memory");
    // P as A-fragment (transpose via LDS)
    bf16x8 plo = *reinterpret_cast<const bf16x8*>(&myP[lc * 40 + kb]);
    bf16x8 phi = *reinterpret_cast<const bf16x8*>(&myP[lc * 40 + kb + 8]);
    bf16x16 aP = cat8(plo, phi);
    // P(16x32) @ V(32x64) from transposed V (contiguous key runs)
#pragma unroll
    for (int nt = 0; nt < 4; nt++) {
      const int hd = nt * 16 + lc;
      const unsigned short* p = &vtws[vtbase + (size_t)hd * Sn + j0 + kscol];
      bf16x8 lo = *reinterpret_cast<const bf16x8*>(p);
      bf16x8 hi = *reinterpret_cast<const bf16x8*>(p + 8);
      bf16x16 bV = cat8(lo, hi);
      accO[nt] = __builtin_amdgcn_wmma_f32_16x16x32_bf16(
          false, aP, false, bV, (short)0, accO[nt], false, false);
    }
  }

  // normalize and store bf16 attention output in [B,S,D] layout
#pragma unroll
  for (int r = 0; r < 8; r++) {
    const float inv = 1.0f / li[r];
    const int row = q0 + r + ((lane >> 4) << 3);
#pragma unroll
    for (int nt = 0; nt < 4; nt++) {
      const int col = h * HDn + nt * 16 + lc;
      aout[((size_t)(b * Sn + row)) * Dn + col] = f2bf_rne(accO[nt][r] * inv);
    }
  }
}

// ---------------------------------------------------------------------------
// Kernel 3: output projection.  attn[8192x1024] (bf16) @ w_proj[1024x1024]
// + b_proj -> out (f32).  Same tiling as kernel 1.
// ---------------------------------------------------------------------------
__global__ __launch_bounds__(256)
void proj_gemm_kernel(const unsigned short* __restrict__ a_bf16,
                      const float* __restrict__ wproj, const float* __restrict__ bproj,
                      float* __restrict__ out) {
  __shared__ unsigned short As[128 * 40];
  __shared__ unsigned short Bt[128 * 40];
  const int tid  = threadIdx.x;
  const int lane = tid & 31;
  const int wave = tid >> 5;
  const int wm = wave & 1, wn = wave >> 1;
  const int m0 = blockIdx.x * 128;
  const int n0 = blockIdx.y * 128;

  f32x8 acc[4][2];
#pragma unroll
  for (int i = 0; i < 4; i++)
#pragma unroll
    for (int j = 0; j < 2; j++) acc[i][j] = zero8();

  const int arow = tid >> 1;          // 0..127
  const int ac16 = (tid & 1) << 4;    // 0 or 16
  const int bcol = tid & 127;
  const int bkh  = (tid >> 7) << 4;

  for (int k0 = 0; k0 < Dn; k0 += 32) {
    __syncthreads();
    {
      const uint4* src = reinterpret_cast<const uint4*>(&a_bf16[(size_t)(m0 + arow) * Dn + k0 + ac16]);
      uint4 v0 = src[0], v1 = src[1];
      uint4* dst = reinterpret_cast<uint4*>(&As[arow * 40 + ac16]);
      dst[0] = v0; dst[1] = v1;
    }
    unsigned short tmp[16];
#pragma unroll
    for (int kr = 0; kr < 16; kr++)
      tmp[kr] = f2bf_rne(wproj[(size_t)(k0 + bkh + kr) * Dn + n0 + bcol]);
#pragma unroll
    for (int q = 0; q < 4; q++) {
      uint2 pk;
      pk.x = (unsigned)tmp[q * 4 + 0] | ((unsigned)tmp[q * 4 + 1] << 16);
      pk.y = (unsigned)tmp[q * 4 + 2] | ((unsigned)tmp[q * 4 + 3] << 16);
      *reinterpret_cast<uint2*>(&Bt[bcol * 40 + bkh + q * 4]) = pk;
    }
    __syncthreads();

    const int kb = (lane >> 4) << 4;
    const int ks = (lane >> 4) << 4;
    bf16x16 afr[4];
#pragma unroll
    for (int mt = 0; mt < 4; mt++) {
      const int row = wm * 64 + mt * 16 + (lane & 15);
      bf16x8 lo = *reinterpret_cast<const bf16x8*>(&As[row * 40 + kb]);
      bf16x8 hi = *reinterpret_cast<const bf16x8*>(&As[row * 40 + kb + 8]);
      afr[mt] = cat8(lo, hi);
    }
    bf16x16 bfr[2];
#pragma unroll
    for (int nt = 0; nt < 2; nt++) {
      const int col = wn * 32 + nt * 16 + (lane & 15);
      bf16x8 lo = *reinterpret_cast<const bf16x8*>(&Bt[col * 40 + ks]);
      bf16x8 hi = *reinterpret_cast<const bf16x8*>(&Bt[col * 40 + ks + 8]);
      bfr[nt] = cat8(lo, hi);
    }
#pragma unroll
    for (int mt = 0; mt < 4; mt++)
#pragma unroll
      for (int nt = 0; nt < 2; nt++)
        acc[mt][nt] = __builtin_amdgcn_wmma_f32_16x16x32_bf16(
            false, afr[mt], false, bfr[nt], (short)0, acc[mt][nt], false, false);
  }

  const int halfm = (lane >> 4) << 3;
#pragma unroll
  for (int nt = 0; nt < 2; nt++) {
    const int Ncol = n0 + wn * 32 + nt * 16 + (lane & 15);
    const float bias = bproj[Ncol];
#pragma unroll
    for (int mt = 0; mt < 4; mt++) {
#pragma unroll
      for (int r = 0; r < 8; r++) {
        const int m = m0 + wm * 64 + mt * 16 + halfm + r;
        out[(size_t)m * Dn + Ncol] = acc[mt][nt][r] + bias;
      }
    }
  }
}

extern "C" void kernel_launch(void* const* d_in, const int* in_sizes, int n_in,
                              void* d_out, int out_size, void* d_ws, size_t ws_size,
                              hipStream_t stream) {
  (void)in_sizes; (void)n_in; (void)out_size; (void)ws_size;
  const float* x      = (const float*)d_in[0];
  const float* w_qkv  = (const float*)d_in[1];
  const float* b_qkv  = (const float*)d_in[2];
  const float* w_proj = (const float*)d_in[3];
  const float* b_proj = (const float*)d_in[4];
  float* out = (float*)d_out;

  const size_t elems = (size_t)Bn * Hn * Sn * HDn;   // 8,388,608
  unsigned short* qws  = (unsigned short*)d_ws;
  unsigned short* kws  = qws + elems;
  unsigned short* vtws = kws + elems;
  unsigned short* aout = vtws + elems;

  // 1) QKV projection (M=8192, N=3072, K=1024), scatter Q/K/Vt bf16
  qkv_gemm_kernel<<<dim3(64, 24), 256, 0, stream>>>(x, w_qkv, b_qkv, qws, kws, vtws);
  // 2) flash attention over (B*H*S/128) = 1024 workgroups
  attn_kernel<<<dim3(1024), 256, 0, stream>>>(qws, kws, vtws, aout);
  // 3) output projection (M=8192, N=1024, K=1024), fp32 out + bias
  proj_gemm_kernel<<<dim3(64, 8), 256, 0, stream>>>(aout, w_proj, b_proj, out);
}
